// Rewire_532575944736
// MI455X (gfx1250) — matrix operand order repaired and checked by reading, not compiled, not run on hardware
//
#include <hip/hip_runtime.h>
#include <math.h>

typedef __attribute__((ext_vector_type(16))) _Float16 v16h;
typedef __attribute__((ext_vector_type(8)))  float    v8f;

#define DF 64   // feature dim

// ---------------------------------------------------------------------------
// 1) Row normalize: mean-center + L2 normalize. One wave (32 lanes) per node,
//    2 features per lane, shuffle reductions (wave32).
// ---------------------------------------------------------------------------
__global__ __launch_bounds__(256) void normalize_h(const float* __restrict__ h,
                                                   float* __restrict__ out, int N) {
    int gid  = blockIdx.x * blockDim.x + threadIdx.x;
    int node = gid >> 5;
    int lane = threadIdx.x & 31;
    if (node >= N) return;
    const float2* hr = (const float2*)(h + (size_t)node * DF);
    float2 v = hr[lane];
    float s = v.x + v.y;
    #pragma unroll
    for (int off = 16; off; off >>= 1) s += __shfl_xor(s, off, 32);
    float mean = s * (1.0f / DF);
    v.x -= mean; v.y -= mean;
    float nn = v.x * v.x + v.y * v.y;
    #pragma unroll
    for (int off = 16; off; off >>= 1) nn += __shfl_xor(nn, off, 32);
    float inv = 1.0f / fmaxf(sqrtf(nn), 1e-12f);
    float2* o = (float2*)(out + (size_t)node * DF);
    o[lane] = make_float2(v.x * inv, v.y * inv);
}

// ---------------------------------------------------------------------------
// 2) k = h @ Wk^T, q = h @ Wq^T via v_wmma_f32_16x16x32_f16.
//    One wave -> 16 rows x 64 cols for BOTH outputs: 2 K-steps x 2 mats x 4
//    col tiles = 16 WMMAs. Per-lane load pattern follows ISA 16-bit A (16x32)
//    and B (32x16) VGPR layouts; C stores follow the 16x16 f32 layout.
// ---------------------------------------------------------------------------
__global__ __launch_bounds__(256) void gemm_kq(const float* __restrict__ h,
                                               const float* __restrict__ Wk,
                                               const float* __restrict__ Wq,
                                               float* __restrict__ kf,
                                               float* __restrict__ qf, int N) {
    int wave = threadIdx.x >> 5;
    int lane = threadIdx.x & 31;
    int i0 = (blockIdx.x * 8 + wave) * 16;
    if (i0 >= N) return;                 // uniform per wave -> EXEC all-ones below
    int half = lane >> 4;                // 0: lanes 0-15, 1: lanes 16-31
    int l16  = lane & 15;
    int r = i0 + l16; if (r >= N) r = N - 1;   // clamp reads; guard stores

    v8f acc[2][4];
    #pragma unroll
    for (int w = 0; w < 2; ++w)
        #pragma unroll
        for (int j = 0; j < 4; ++j)
            #pragma unroll
            for (int t = 0; t < 8; ++t) acc[w][j][t] = 0.0f;

    const float* WW0 = Wk;
    const float* WW1 = Wq;

    #pragma unroll
    for (int kk0 = 0; kk0 < DF; kk0 += 32) {
        // A tile (16x32 f16): lane<16 -> row l16, K in {kk0+0..7, kk0+16..23}
        //                     lane>=16 -> row l16, K shifted by +8
        int abase = kk0 + half * 8;
        const float* hr = h + (size_t)r * DF + abase;
        v16h a;
        #pragma unroll
        for (int t = 0; t < 8; ++t) a[t]     = (_Float16)hr[t];
        #pragma unroll
        for (int t = 0; t < 8; ++t) a[8 + t] = (_Float16)hr[16 + t];

        int bbase = kk0 + half * 8;      // B: same K split as A (+8 for lanes>=16)
        #pragma unroll
        for (int w = 0; w < 2; ++w) {
            const float* W = (w == 0) ? WW0 : WW1;
            #pragma unroll
            for (int j = 0; j < 4; ++j) {
                // B[kk, n] = W[j*16 + n, kk]  (col n = l16), K {+0..7, +16..23}
                const float* wr = W + (size_t)(j * 16 + l16) * DF + bbase;
                v16h b;
                #pragma unroll
                for (int t = 0; t < 8; ++t) { b[t] = (_Float16)wr[t]; b[8 + t] = (_Float16)wr[16 + t]; }
                acc[w][j] = __builtin_amdgcn_wmma_f32_16x16x32_f16(
                    false, a, false, b, (short)0, acc[w][j], false, false);
            }
        }
    }

    #pragma unroll
    for (int w = 0; w < 2; ++w) {
        float* out = (w == 0) ? kf : qf;
        #pragma unroll
        for (int j = 0; j < 4; ++j)
            #pragma unroll
            for (int v = 0; v < 8; ++v) {
                int row = i0 + v + half * 8;   // C layout: lanes 16-31 hold M=v+8
                if (row < N) out[(size_t)row * DF + j * 16 + l16] = acc[w][j][v];
            }
    }
}

// ---------------------------------------------------------------------------
// 3) Segment softmax over incoming edges of each dst.
// ---------------------------------------------------------------------------
__device__ __forceinline__ void atomicMaxF(float* addr, float value) {
    float old = __int_as_float(__ldg((const int*)addr) ^ 0);  // plain load
    old = *addr;
    while (old < value) {
        int assumed = __float_as_int(old);
        int prev = atomicCAS((int*)addr, assumed, __float_as_int(value));
        if (prev == assumed) break;
        old = __int_as_float(prev);
    }
}

__global__ __launch_bounds__(256) void init_ms(float* __restrict__ m,
                                               float* __restrict__ s, int N) {
    int i = blockIdx.x * blockDim.x + threadIdx.x;
    if (i < N) { m[i] = -__builtin_inff(); s[i] = 0.0f; }
}

__global__ __launch_bounds__(256) void edge_scores(const float* __restrict__ kf,
                                                   const float* __restrict__ qf,
                                                   const int* __restrict__ src,
                                                   const int* __restrict__ dst,
                                                   float* __restrict__ e,
                                                   float* __restrict__ m, int E) {
    int i = blockIdx.x * blockDim.x + threadIdx.x;
    if (i >= E) return;
    int s = src[i], d = dst[i];
    const float4* ks = (const float4*)(kf + (size_t)s * DF);
    const float4* qd = (const float4*)(qf + (size_t)d * DF);
    const float4* qs = (const float4*)(qf + (size_t)s * DF);
    const float4* kd = (const float4*)(kf + (size_t)d * DF);
    float acc = 0.0f;
    #pragma unroll
    for (int t = 0; t < DF / 4; ++t) {
        float4 a = ks[t], b = qd[t];
        acc += a.x * b.x + a.y * b.y + a.z * b.z + a.w * b.w;
        float4 c = qs[t], g = kd[t];
        acc += c.x * g.x + c.y * g.y + c.z * g.z + c.w * g.w;
    }
    float ev = acc * 0.125f;             // 1/sqrt(64)
    e[i] = ev;
    atomicMaxF(m + d, ev);
}

__global__ __launch_bounds__(256) void edge_exp(const float* __restrict__ m,
                                                const int* __restrict__ dst,
                                                float* __restrict__ e,
                                                float* __restrict__ s, int E) {
    int i = blockIdx.x * blockDim.x + threadIdx.x;
    if (i >= E) return;
    int d = dst[i];
    float mm = m[d];
    if (!(mm > -3.0e38f)) mm = 0.0f;     // isfinite guard (m init = -inf)
    float ex = __expf(e[i] - mm);
    e[i] = ex;
    atomicAdd(s + d, ex);
}

__global__ __launch_bounds__(256) void edge_norm(const float* __restrict__ s,
                                                 const int* __restrict__ dst,
                                                 float* __restrict__ e, int E) {
    int i = blockIdx.x * blockDim.x + threadIdx.x;
    if (i >= E) return;
    e[i] = e[i] / s[dst[i]];
}

// ---------------------------------------------------------------------------
// 4) ODE stages
// ---------------------------------------------------------------------------
__global__ __launch_bounds__(256) void zero_f4(float4* __restrict__ p, int n4) {
    int i = blockIdx.x * blockDim.x + threadIdx.x;
    if (i < n4) p[i] = make_float4(0.f, 0.f, 0.f, 0.f);
}

// agg[dst] += w_e * hh[src] : one wave per edge, float2 per lane, f32 atomics.
__global__ __launch_bounds__(256) void spmm_scatter(const float* __restrict__ w,
                                                    const int* __restrict__ src,
                                                    const int* __restrict__ dst,
                                                    const float* __restrict__ hh,
                                                    float* __restrict__ agg, int E) {
    int gid  = blockIdx.x * blockDim.x + threadIdx.x;
    int edge = gid >> 5;
    int lane = threadIdx.x & 31;
    if (edge >= E) return;
    int s = src[edge], d = dst[edge];
    float we = w[edge];
    float2 v = ((const float2*)(hh + (size_t)s * DF))[lane];
    float* base = agg + (size_t)d * DF + lane * 2;
    atomicAdd(base + 0, we * v.x);
    atomicAdd(base + 1, we * v.y);
}

// kj = sig(alpha)*(agg - hh_in);  hnext = hbase + dt*(c1*k1+..+c5*k5 + cj*kj)
__global__ __launch_bounds__(256) void stage_combine(
        const float4* __restrict__ agg, const float4* __restrict__ hh_in,
        float4* __restrict__ kout,
        const float4* __restrict__ hbase, float4* __restrict__ hnext,
        const float4* __restrict__ k1, const float4* __restrict__ k2,
        const float4* __restrict__ k3, const float4* __restrict__ k4,
        const float4* __restrict__ k5,
        float c1, float c2, float c3, float c4, float c5, float cj,
        const float* __restrict__ alpha, float dt, int n4) {
    int i = blockIdx.x * blockDim.x + threadIdx.x;
    if (i >= n4) return;
    float sg = 1.0f / (1.0f + __expf(-alpha[0]));
    float4 a = agg[i], hi = hh_in[i];
    float4 kj = make_float4(sg * (a.x - hi.x), sg * (a.y - hi.y),
                            sg * (a.z - hi.z), sg * (a.w - hi.w));
    kout[i] = kj;
    float4 acc = make_float4(cj * kj.x, cj * kj.y, cj * kj.z, cj * kj.w);
    if (c1 != 0.f) { float4 t = k1[i]; acc.x += c1*t.x; acc.y += c1*t.y; acc.z += c1*t.z; acc.w += c1*t.w; }
    if (c2 != 0.f) { float4 t = k2[i]; acc.x += c2*t.x; acc.y += c2*t.y; acc.z += c2*t.z; acc.w += c2*t.w; }
    if (c3 != 0.f) { float4 t = k3[i]; acc.x += c3*t.x; acc.y += c3*t.y; acc.z += c3*t.z; acc.w += c3*t.w; }
    if (c4 != 0.f) { float4 t = k4[i]; acc.x += c4*t.x; acc.y += c4*t.y; acc.z += c4*t.z; acc.w += c4*t.w; }
    if (c5 != 0.f) { float4 t = k5[i]; acc.x += c5*t.x; acc.y += c5*t.y; acc.z += c5*t.z; acc.w += c5*t.w; }
    float4 hb = hbase[i];
    hnext[i] = make_float4(hb.x + dt * acc.x, hb.y + dt * acc.y,
                           hb.z + dt * acc.z, hb.w + dt * acc.w);
}

// ---------------------------------------------------------------------------
extern "C" void kernel_launch(void* const* d_in, const int* in_sizes, int n_in,
                              void* d_out, int out_size, void* d_ws, size_t ws_size,
                              hipStream_t stream) {
    const float* h     = (const float*)d_in[0];
    const int*   src   = (const int*)  d_in[1];
    const int*   dst   = (const int*)  d_in[2];
    const float* Wk    = (const float*)d_in[3];
    const float* Wq    = (const float*)d_in[4];
    const float* alpha = (const float*)d_in[5];
    int N = in_sizes[0] / DF;
    int E = in_sizes[1];
    float* out = (float*)d_out;
    size_t ND = (size_t)N * DF;

    // workspace layout (floats): e[E], m[N], s[N], 8 x ND buffers.
    float* wsf   = (float*)d_ws;
    float* e_buf = wsf;
    float* mbuf  = e_buf + E;
    float* sbuf  = mbuf + N;
    float* buf   = sbuf + N;
    float* kf = buf;               // aliases kb[0] (free after edge scoring)
    float* qf = buf + ND;          // aliases kb[1]
    float* kb[6];
    for (int i = 0; i < 6; ++i) kb[i] = buf + (size_t)i * ND;
    float* hh_tmp = buf + 6 * ND;
    float* agg    = buf + 7 * ND;

    float* h0 = out;               // slice 0 = normalized h

    const int TB = 256;
    // 1) normalize
    normalize_h<<<(int)(((size_t)N * 32 + TB - 1) / TB), TB, 0, stream>>>(h, h0, N);
    // 2) k/q projection (WMMA)
    gemm_kq<<<(N + 127) / 128, TB, 0, stream>>>(h0, Wk, Wq, kf, qf, N);
    // 3) edge softmax
    init_ms<<<(N + TB - 1) / TB, TB, 0, stream>>>(mbuf, sbuf, N);
    edge_scores<<<(E + TB - 1) / TB, TB, 0, stream>>>(kf, qf, src, dst, e_buf, mbuf, E);
    edge_exp<<<(E + TB - 1) / TB, TB, 0, stream>>>(mbuf, dst, e_buf, sbuf, E);
    edge_norm<<<(E + TB - 1) / TB, TB, 0, stream>>>(sbuf, dst, e_buf, E);

    // 4) DoPri5: 7 steps, 6 stages each. Coeffs to build next stage input
    //    after computing stage j: {c1..c5, cj}
    static const float C[6][6] = {
        {0.f, 0.f, 0.f, 0.f, 0.f, 1.f/5.f},
        {3.f/40.f, 0.f, 0.f, 0.f, 0.f, 9.f/40.f},
        {44.f/45.f, -56.f/15.f, 0.f, 0.f, 0.f, 32.f/9.f},
        {19372.f/6561.f, -25360.f/2187.f, 64448.f/6561.f, 0.f, 0.f, -212.f/729.f},
        {9017.f/3168.f, -355.f/33.f, 46732.f/5247.f, 49.f/176.f, 0.f, -5103.f/18656.f},
        {35.f/384.f, 0.f, 500.f/1113.f, 125.f/192.f, -2187.f/6784.f, 11.f/84.f},
    };
    const float dt = 1.0f / 7.0f;   // T/(NT-1)
    int n4 = (int)(ND / 4);
    int gZero    = (n4 + TB - 1) / TB;
    int gScatter = (int)(((size_t)E * 32 + TB - 1) / TB);

    for (int step = 0; step < 7; ++step) {
        const float* hb = out + (size_t)step * ND;
        float*       hn = out + (size_t)(step + 1) * ND;
        for (int st = 0; st < 6; ++st) {
            const float* hin = (st == 0) ? hb : hh_tmp;
            float* hnext = (st == 5) ? hn : hh_tmp;
            zero_f4<<<gZero, TB, 0, stream>>>((float4*)agg, n4);
            spmm_scatter<<<gScatter, TB, 0, stream>>>(e_buf, src, dst, hin, agg, E);
            stage_combine<<<gZero, TB, 0, stream>>>(
                (const float4*)agg, (const float4*)hin, (float4*)kb[st],
                (const float4*)hb, (float4*)hnext,
                (const float4*)kb[0], (const float4*)kb[1], (const float4*)kb[2],
                (const float4*)kb[3], (const float4*)kb[4],
                C[st][0], C[st][1], C[st][2], C[st][3], C[st][4], C[st][5],
                alpha, dt, n4);
        }
    }
}
